// BloomAttention_36197984371380
// MI455X (gfx1250) — matrix operand order repaired and checked by reading, not compiled, not run on hardware
//
#include <hip/hip_runtime.h>
#include <stdint.h>

typedef __bf16 bf16;
typedef __attribute__((ext_vector_type(16))) __bf16 v16bf;
typedef __attribute__((ext_vector_type(4)))  __bf16 v4bf;
typedef __attribute__((ext_vector_type(8)))  float  v8f;
typedef __attribute__((ext_vector_type(4)))  float  v4f;
typedef __attribute__((ext_vector_type(4)))  unsigned int u32x4;
typedef __attribute__((ext_vector_type(4)))  int v4i;

#define B_DIM 2
#define S_LEN 2048
#define H_DIM 4096
#define NH 32
#define HD 128
#define INV_NORM 0.08838834764831845f /* 1/sqrt(128) */
#define MASK_VAL -1e9f

// ---- optional gfx1250 async-to-LDS path (probe-guarded) ----
#if defined(__has_builtin)
#if __has_builtin(__builtin_amdgcn_global_load_async_to_lds_b128) && \
    __has_builtin(__builtin_amdgcn_s_wait_asynccnt)
#define USE_ASYNC_LDS 1
#endif
#endif

#ifdef USE_ASYNC_LDS
typedef __attribute__((address_space(1))) v4i g_v4i;
typedef __attribute__((address_space(3))) v4i l_v4i;
static __device__ __forceinline__ g_v4i* to_glob128(const void* p) {
    return (g_v4i*)(uintptr_t)p;                 // generic global addr == AS1 addr
}
static __device__ __forceinline__ l_v4i* to_lds128(void* p) {
    // generic LDS address: [63:32]=shared aperture, [31:0]=LDS offset
    return (l_v4i*)(unsigned)(uintptr_t)p;
}
#endif

// ---------- WMMA helper ----------
static __device__ __forceinline__ v8f wmma_bf16(v16bf a, v16bf b, v8f c) {
    return __builtin_amdgcn_wmma_f32_16x16x32_bf16(false, a, false, b, (short)0, c,
                                                   false, false);
}

// A fragment (16x32 bf16, row-major source, 16-byte aligned):
// lane l: row = l%16, K = {0..7,16..23} (l<16) or {8..15,24..31} (l>=16)
static __device__ __forceinline__ v16bf load_a_frag(const bf16* base, int ld) {
    int lane = threadIdx.x & 31;
    const bf16* p = base + (size_t)(lane & 15) * ld + ((lane >> 4) << 3);
    union { v16bf v; u32x4 u[2]; } r;
    r.u[0] = *(const u32x4*)(p);
    r.u[1] = *(const u32x4*)(p + 16);
    return r.v;
}

// B fragment (32x16 bf16) from an "n-major" matrix Bsrc[n][k]:
// lane l: col n = l%16, K = {0..7,16..23} (l<16) or {8..15,24..31} (l>=16)
static __device__ __forceinline__ v16bf load_b_frag(const bf16* base, int ld) {
    int lane = threadIdx.x & 31;
    const bf16* p = base + (size_t)(lane & 15) * ld + ((lane >> 4) << 3);
    union { v16bf v; u32x4 u[2]; } r;
    r.u[0] = *(const u32x4*)(p);
    r.u[1] = *(const u32x4*)(p + 16);
    return r.v;
}

// Cooperative fill of a 64(n) x 32(k) bf16 LDS panel from an n-major matrix.
// 128 threads x 2 chunks of 16B each.
static __device__ __forceinline__ void fill_b_panel(bf16* dst, const bf16* Brow,
                                                    int k0) {
    int tid = threadIdx.y * 32 + (threadIdx.x & 31);
    #pragma unroll
    for (int j = 0; j < 2; ++j) {
        int ci  = tid * 2 + j;          // 0..255
        int row = ci >> 2;              // 0..63
        int kc  = (ci & 3) * 8;         // 0,8,16,24
        const bf16* g = Brow + (size_t)row * H_DIM + k0 + kc;
        bf16* l = dst + row * 32 + kc;
#ifdef USE_ASYNC_LDS
        __builtin_amdgcn_global_load_async_to_lds_b128(to_glob128(g), to_lds128(l), 0, 0);
#else
        *(u32x4*)l = *(const u32x4*)g;
#endif
    }
}

static __device__ __forceinline__ void fill_wait() {
#ifdef USE_ASYNC_LDS
    __builtin_amdgcn_s_wait_asynccnt(0);
#endif
}

// ---------- fp32 -> bf16 convert (vectorized: b128 load, b64 store) ----------
__global__ void bloom_cvt_bf16(const float* __restrict__ in, bf16* __restrict__ out, int n4) {
    int i = blockIdx.x * blockDim.x + threadIdx.x;
    if (i < n4) {
        v4f f = *(const v4f*)(in + (size_t)i * 4);
        v4bf o;
        o[0] = (bf16)f[0]; o[1] = (bf16)f[1]; o[2] = (bf16)f[2]; o[3] = (bf16)f[3];
        *(v4bf*)(out + (size_t)i * 4) = o;
    }
}

// ---------- fused QKV GEMM: fused = X @ Wqkv^T + b, scattered to Q/K/Vt ----------
// Block: 4 waves = 64(M) x 64(N) tile; B panel staged in LDS (double-buffered,
// async when available); each wave reuses one A fragment across 4 WMMAs.
__global__ __launch_bounds__(128) void bloom_qkv_wmma(const bf16* __restrict__ X,
                                                      const bf16* __restrict__ W,
                                                      const float* __restrict__ bias,
                                                      bf16* __restrict__ Q,
                                                      bf16* __restrict__ K,
                                                      bf16* __restrict__ Vt) {
    const int n0 = blockIdx.x * 64;                         // 0..12224
    const int m0 = (blockIdx.y * 4 + threadIdx.y) * 16;     // 0..4080 (token)
    const int lane = threadIdx.x & 31;
    const int nc = lane & 15, half = lane >> 4;

    const bf16* Arow = X + (size_t)m0 * H_DIM;
    const bf16* Brow = W + (size_t)n0 * H_DIM;

    __shared__ alignas(16) bf16 Bsh[2][64 * 32];

    v8f acc[4] = {{}, {}, {}, {}};

    fill_b_panel(Bsh[0], Brow, 0);
    fill_wait();
    __syncthreads();

    const int KT = H_DIM / 32;
    for (int kb = 0; kb < KT; ++kb) {
        const int cur = kb & 1;
        if (kb + 1 < KT) fill_b_panel(Bsh[1 - cur], Brow, (kb + 1) * 32);
        __builtin_prefetch(Arow + (kb + 1) * 32, 0, 1);
        v16bf a = load_a_frag(Arow + kb * 32, H_DIM);
        #pragma unroll
        for (int t = 0; t < 4; ++t) {
            v16bf b = load_b_frag(&Bsh[cur][t * 16 * 32], 32);
            acc[t] = wmma_bf16(a, b, acc[t]);
        }
        fill_wait();
        __syncthreads();
    }

    #pragma unroll
    for (int t = 0; t < 4; ++t) {
        const int n = n0 + t * 16 + nc;
        const float bv = bias[n];
        const int nh = n / (3 * HD);
        const int r  = n % (3 * HD);
        const int part = r / HD;
        const int hd   = r % HD;
        #pragma unroll
        for (int v = 0; v < 8; ++v) {
            int tok = m0 + v + 8 * half;
            int bb  = tok >> 11;               // /S_LEN
            int ss  = tok & (S_LEN - 1);
            bf16 val = (bf16)(acc[t][v] + bv);
            size_t hb = (size_t)(bb * NH + nh);
            if (part == 0)      Q[(hb * S_LEN + ss) * HD + hd] = val;
            else if (part == 1) K[(hb * S_LEN + ss) * HD + hd] = val;
            else                Vt[(hb * HD + hd) * S_LEN + ss] = val;
        }
    }
}

// ---------- flash attention with ALiBi + causal mask ----------
// grid: (B*NH, S/16); one wave per 16-query tile
__global__ __launch_bounds__(32) void bloom_attn_wmma(const bf16* __restrict__ Q,
                                                      const bf16* __restrict__ K,
                                                      const bf16* __restrict__ Vt,
                                                      const float* __restrict__ alibi,
                                                      bf16* __restrict__ ctx_out) {
    const int bh = blockIdx.x;            // 0..63
    const int q0 = blockIdx.y * 16;
    const int b  = bh / NH, nh = bh % NH;
    const int lane = threadIdx.x & 31;
    const int nc = lane & 15, half = lane >> 4;

    const bf16* Qb = Q + ((size_t)bh * S_LEN + q0) * HD;
    const bf16* Kb = K + (size_t)bh * S_LEN * HD;
    const bf16* Vb = Vt + (size_t)bh * HD * S_LEN;
    const float* al = alibi + (size_t)bh * S_LEN;

    // Q fragments over HD=128 (4 x K=32), kept in registers
    v16bf qf[4];
    #pragma unroll
    for (int i = 0; i < 4; ++i) qf[i] = load_a_frag(Qb + i * 32, HD);

    v8f ctx[8];
    #pragma unroll
    for (int t = 0; t < 8; ++t) ctx[t] = (v8f){};
    float run_max[8], run_sum[8];
    #pragma unroll
    for (int v = 0; v < 8; ++v) { run_max[v] = -3.0e38f; run_sum[v] = 0.0f; }

    __shared__ alignas(16) bf16 pbuf[16 * 32];

    const int kb_end = (q0 + 15) / 32;    // causal: last key block touched
    for (int kb = 0; kb <= kb_end; ++kb) {
        const int k0 = kb * 32;
        __builtin_prefetch(Kb + (size_t)(k0 + 32) * HD, 0, 1);
        v8f s0 = {}, s1 = {};
        #pragma unroll
        for (int i = 0; i < 4; ++i) {
            v16bf b0 = load_b_frag(Kb + (size_t)k0 * HD + i * 32, HD);
            v16bf b1 = load_b_frag(Kb + (size_t)(k0 + 16) * HD + i * 32, HD);
            s0 = wmma_bf16(qf[i], b0, s0);
            s1 = wmma_bf16(qf[i], b1, s1);
        }

        const float a0 = al[k0 + nc];
        const float a1 = al[k0 + 16 + nc];
        float p0[8], p1[8];
        #pragma unroll
        for (int v = 0; v < 8; ++v) {
            int qrow = q0 + v + 8 * half;
            p0[v] = s0[v] * INV_NORM + a0 + ((k0 + nc)      > qrow ? MASK_VAL : 0.0f);
            p1[v] = s1[v] * INV_NORM + a1 + ((k0 + 16 + nc) > qrow ? MASK_VAL : 0.0f);
        }

        // online softmax: per-row (v index) stats across the 16-lane half
        #pragma unroll
        for (int v = 0; v < 8; ++v) {
            float m = fmaxf(p0[v], p1[v]);
            #pragma unroll
            for (int off = 1; off < 16; off <<= 1) m = fmaxf(m, __shfl_xor(m, off, 32));
            float nm = fmaxf(run_max[v], m);
            float corr = __expf(run_max[v] - nm);
            run_max[v] = nm;
            float e0 = __expf(p0[v] - nm);
            float e1 = __expf(p1[v] - nm);
            p0[v] = e0; p1[v] = e1;
            float rs = e0 + e1;
            #pragma unroll
            for (int off = 1; off < 16; off <<= 1) rs += __shfl_xor(rs, off, 32);
            run_sum[v] = run_sum[v] * corr + rs;
            #pragma unroll
            for (int t = 0; t < 8; ++t) ctx[t][v] *= corr;
        }

        // repack P (C-layout f32 -> A-layout bf16) through LDS
        #pragma unroll
        for (int v = 0; v < 8; ++v) {
            int m = v + 8 * half;
            pbuf[m * 32 + nc]      = (bf16)p0[v];
            pbuf[m * 32 + 16 + nc] = (bf16)p1[v];
        }
        v16bf pf = load_a_frag(pbuf, 32);   // same-wave DS ordering is in-order

        // ctx += P @ V  (V transposed: B(k,n)=Vt[hd=n][key=k], contiguous in k)
        #pragma unroll
        for (int t = 0; t < 8; ++t) {
            v16bf vf = load_b_frag(Vb + (size_t)(t * 16) * S_LEN + k0, S_LEN);
            ctx[t] = wmma_bf16(pf, vf, ctx[t]);
        }
    }

    // normalize + merge heads: ctx_out [B*S, H] bf16
    #pragma unroll
    for (int t = 0; t < 8; ++t) {
        #pragma unroll
        for (int v = 0; v < 8; ++v) {
            int m = v + 8 * half;
            size_t tok = (size_t)(b * S_LEN + q0 + m);
            ctx_out[tok * H_DIM + nh * HD + t * 16 + nc] = (bf16)(ctx[t][v] / run_sum[v]);
        }
    }
}

// ---------- dense GEMM + bias + residual -> fp32 out ----------
// Block: 4 waves = 64(M) x 64(N) tile; LDS-staged B panel like QKV.
__global__ __launch_bounds__(128) void bloom_dense_wmma(const bf16* __restrict__ Ctx,
                                                        const bf16* __restrict__ W,
                                                        const float* __restrict__ bias,
                                                        const float* __restrict__ resid,
                                                        float* __restrict__ out) {
    const int n0 = blockIdx.x * 64;
    const int m0 = (blockIdx.y * 4 + threadIdx.y) * 16;
    const int lane = threadIdx.x & 31;
    const int nc = lane & 15, half = lane >> 4;

    const bf16* Arow = Ctx + (size_t)m0 * H_DIM;
    const bf16* Brow = W + (size_t)n0 * H_DIM;

    __shared__ alignas(16) bf16 Bsh[2][64 * 32];

    v8f acc[4] = {{}, {}, {}, {}};

    fill_b_panel(Bsh[0], Brow, 0);
    fill_wait();
    __syncthreads();

    const int KT = H_DIM / 32;
    for (int kb = 0; kb < KT; ++kb) {
        const int cur = kb & 1;
        if (kb + 1 < KT) fill_b_panel(Bsh[1 - cur], Brow, (kb + 1) * 32);
        __builtin_prefetch(Arow + (kb + 1) * 32, 0, 1);
        v16bf a = load_a_frag(Arow + kb * 32, H_DIM);
        #pragma unroll
        for (int t = 0; t < 4; ++t) {
            v16bf b = load_b_frag(&Bsh[cur][t * 16 * 32], 32);
            acc[t] = wmma_bf16(a, b, acc[t]);
        }
        fill_wait();
        __syncthreads();
    }

    #pragma unroll
    for (int t = 0; t < 4; ++t) {
        const int n = n0 + t * 16 + nc;
        const float bv = bias[n];
        #pragma unroll
        for (int v = 0; v < 8; ++v) {
            size_t tok = (size_t)(m0 + v + 8 * half);
            out[tok * H_DIM + n] = acc[t][v] + bv + resid[tok * H_DIM + n];
        }
    }
}

extern "C" void kernel_launch(void* const* d_in, const int* in_sizes, int n_in,
                              void* d_out, int out_size, void* d_ws, size_t ws_size,
                              hipStream_t stream) {
    const float* hidden   = (const float*)d_in[0];
    const float* residual = (const float*)d_in[1];
    const float* alibi    = (const float*)d_in[2];
    // d_in[3] attention_mask: causal mask recomputed analytically
    const float* W_qkv    = (const float*)d_in[4];
    const float* b_qkv    = (const float*)d_in[5];
    const float* W_dense  = (const float*)d_in[6];
    const float* b_dense  = (const float*)d_in[7];
    float* out = (float*)d_out;

    const size_t N_HS  = (size_t)B_DIM * S_LEN * H_DIM;   // 16,777,216
    const size_t N_WQ  = (size_t)3 * H_DIM * H_DIM;       // 50,331,648
    const size_t N_WD  = (size_t)H_DIM * H_DIM;           // 16,777,216
    const size_t N_QKV = (size_t)B_DIM * NH * S_LEN * HD; // 16,777,216

    char* ws = (char*)d_ws;
    size_t off = 0;
    bf16* hsb   = (bf16*)(ws + off); off += N_HS  * 2;
    bf16* wqkvb = (bf16*)(ws + off); off += N_WQ  * 2;
    bf16* wdb   = (bf16*)(ws + off); off += N_WD  * 2;
    bf16* Qb    = (bf16*)(ws + off); off += N_QKV * 2;
    bf16* Kb    = (bf16*)(ws + off); off += N_QKV * 2;
    bf16* Vtb   = (bf16*)(ws + off); off += N_QKV * 2;
    bf16* ctxb  = (bf16*)(ws + off); off += N_HS  * 2;

    // 1) fp32 -> bf16 staging (vectorized x4; all sizes are multiples of 4)
    bloom_cvt_bf16<<<(int)((N_HS / 4 + 255) / 256), 256, 0, stream>>>(hidden,  hsb,   (int)(N_HS / 4));
    bloom_cvt_bf16<<<(int)((N_WQ / 4 + 255) / 256), 256, 0, stream>>>(W_qkv,   wqkvb, (int)(N_WQ / 4));
    bloom_cvt_bf16<<<(int)((N_WD / 4 + 255) / 256), 256, 0, stream>>>(W_dense, wdb,   (int)(N_WD / 4));

    // 2) fused QKV projection (WMMA, LDS-staged B, 64x64 per block)
    bloom_qkv_wmma<<<dim3(3 * H_DIM / 64, (B_DIM * S_LEN) / 64), dim3(32, 4), 0, stream>>>(
        hsb, wqkvb, b_qkv, Qb, Kb, Vtb);

    // 3) flash attention with ALiBi + causal (WMMA)
    bloom_attn_wmma<<<dim3(B_DIM * NH, S_LEN / 16), 32, 0, stream>>>(
        Qb, Kb, Vtb, alibi, ctxb);

    // 4) dense projection + bias + residual (WMMA, LDS-staged B, 64x64 per block)
    bloom_dense_wmma<<<dim3(H_DIM / 64, (B_DIM * S_LEN) / 64), dim3(32, 4), 0, stream>>>(
        ctxb, wdb, b_dense, residual, out);
}
